// CurvphormerLayer_4346506903649
// MI455X (gfx1250) — matrix operand order repaired and checked by reading, not compiled, not run on hardware
//
#include <hip/hip_runtime.h>

#define NN   50000
#define EE   800000
#define HH   128
#define NHH  8
#define NT   3125          // 16-row tiles
#define NB   782           // blocks of 4 tiles (782*4 = 3128 >= 3125)

typedef __attribute__((ext_vector_type(16))) __bf16          v16bf;
typedef __attribute__((ext_vector_type(16))) unsigned short  v16us;
typedef __attribute__((ext_vector_type(8)))  float           v8f;

// ---------- bf16 helpers (native converts) ----------
__device__ __forceinline__ unsigned short f2bf(float f) {
    __bf16 h = (__bf16)f;
    return __builtin_bit_cast(unsigned short, h);
}
__device__ __forceinline__ float bf2f(unsigned short u) {
    return (float)__builtin_bit_cast(__bf16, u);
}

// ---------- CDNA5 async global -> LDS copy (ASYNCcnt-tracked) ----------
// GV mode: VDST = LDS byte address, VADDR = 64-bit global address, SADDR = off.
// Low 32 bits of a flat shared pointer are the LDS byte offset.
__device__ __forceinline__ void async_copy16(void* lds_dst, const void* gsrc) {
    unsigned lds_off = (unsigned)(uintptr_t)lds_dst;
    asm volatile("global_load_async_to_lds_b128 %0, %1, off"
                 :: "v"(lds_off), "v"(gsrc)
                 : "memory");
}
__device__ __forceinline__ void async_wait0() {
    asm volatile("s_wait_asynccnt 0x0" ::: "memory");
}

// ---------- WMMA fragment builders (CDNA5 16x16x32 bf16 layouts) ----------
// A (16x32 MxK): lane<16 rows 0..15 K-sub {0..7,16..23}; lane>=16 K+{8,24}.
// Each half-fragment is 8 contiguous bf16 -> two 16B LDS loads.
__device__ __forceinline__ v16bf a_frag_lds(const unsigned short* t, int ld, int k0, int lane) {
    int row = lane & 15, half = lane >> 4;
    const unsigned short* p = t + row * ld + k0 + half * 8;
    uint4 lo = *(const uint4*)p;
    uint4 hi = *(const uint4*)(p + 16);
    union { uint4 q[2]; v16us v; } u;
    u.q[0] = lo; u.q[1] = hi;
    return __builtin_bit_cast(v16bf, u.v);
}
__device__ __forceinline__ v16bf a_frag_gf32(const float* base, int ld, int k0, int lane) {
    int row = lane & 15, half = lane >> 4;
    const float* r0 = base + (size_t)row * ld + k0 + half * 8;
    const float* r1 = r0 + 16;
    v16bf v;
#pragma unroll
    for (int j = 0; j < 8; ++j) { v[j] = (__bf16)r0[j]; v[8 + j] = (__bf16)r1[j]; }
    return v;
}
// B (32x16 KxN): lanes 0-15 K {0..7,16..23}, lanes 16-31 K {8..15,24..31} (matches A); N=lane&15
__device__ __forceinline__ v16bf b_frag_gf32(const float* W, int ldw, int k0, int c0, int lane, int maxcol) {
    int col = lane & 15;
    int kh  = (lane >> 4) << 3;
    bool ok = (c0 + col) < maxcol;
    v16bf v;
#pragma unroll
    for (int i = 0; i < 8; ++i) {
        int kk = k0 + kh + 2 * i + (i >= 4 ? 8 : 0);
        v[2 * i]     = (__bf16)(ok ? W[(size_t)kk * ldw + c0 + col] : 0.f);
        v[2 * i + 1] = (__bf16)(ok ? W[(size_t)(kk + 1) * ldw + c0 + col] : 0.f);
    }
    return v;
}
__device__ __forceinline__ v8f wmma_bf16(v16bf a, v16bf b, v8f c) {
    return __builtin_amdgcn_wmma_f32_16x16x32_bf16(false, a, false, b, (short)0, c, false, false);
}

// ---------- float atomic max via CAS ----------
__device__ __forceinline__ void atomicMaxF(float* a, float v) {
    int cur = __float_as_int(*((volatile float*)a));
    while (v > __int_as_float(cur)) {
        int prev = atomicCAS((int*)a, cur, __float_as_int(v));
        if (prev == cur) break;
        cur = prev;
    }
}

// ---------- layernorm of one 16x128 LDS tile -> bf16 tile ----------
__device__ __forceinline__ void ln_tile(const float* xf, unsigned short* out,
                                        const float* g, const float* bta,
                                        float* red, float* musig, int tid) {
    int row = tid >> 4, sub = tid & 15;
    float s = 0.f, s2 = 0.f;
#pragma unroll
    for (int j = 0; j < 8; ++j) {
        float v = xf[row * 128 + sub * 8 + j];
        s += v; s2 += v * v;
    }
    red[tid * 2 + 0] = s;
    red[tid * 2 + 1] = s2;
    __syncthreads();
    if (tid < 16) {
        float ts = 0.f, ts2 = 0.f;
#pragma unroll
        for (int i = 0; i < 16; ++i) { ts += red[(tid * 16 + i) * 2]; ts2 += red[(tid * 16 + i) * 2 + 1]; }
        float mu  = ts * (1.f / 128.f);
        float var = ts2 * (1.f / 128.f) - mu * mu;
        musig[tid * 2]     = mu;
        musig[tid * 2 + 1] = rsqrtf(var + 1e-5f);
    }
    __syncthreads();
    float mu = musig[row * 2], rs = musig[row * 2 + 1];
#pragma unroll
    for (int j = 0; j < 8; ++j) {
        int c = sub * 8 + j;
        float v = xf[row * 128 + c];
        out[row * 128 + c] = f2bf((v - mu) * rs * g[c] + bta[c]);
    }
    __syncthreads();
}

// ============ K0: init accumulators ============
__global__ __launch_bounds__(256) void k0_init(float* smax, float* ssum, float* outacc) {
    size_t i = (size_t)blockIdx.x * 256 + threadIdx.x;
    if (i < (size_t)NN * HH) outacc[i] = 0.f;
    if (i < (size_t)NN * NHH) { smax[i] = -1e30f; ssum[i] = 0.f; }
}

// ============ K1: LN1 + Q/K/V projection, 4 row-tiles/block, hoisted B ============
__global__ __launch_bounds__(256) void k1_ln_qkv(
    const float* __restrict__ x,
    const float* __restrict__ Wq, const float* __restrict__ bq,
    const float* __restrict__ Wk, const float* __restrict__ bk,
    const float* __restrict__ Wv, const float* __restrict__ bv,
    const float* __restrict__ g1, const float* __restrict__ be1,
    unsigned short* __restrict__ qo, unsigned short* __restrict__ ko, unsigned short* __restrict__ vo) {
    __shared__ __align__(16) float xf[2048];
    __shared__ __align__(16) unsigned short xbf[2048];
    __shared__ float red[512];
    __shared__ float musig[32];
    int tid = threadIdx.x, wave = tid >> 5, lane = tid & 31;
    int col = lane & 15, half = lane >> 4, c0 = wave * 16;

    v16bf Bq[4], Bk[4], Bv[4];
#pragma unroll
    for (int ks = 0; ks < 4; ++ks) {
        Bq[ks] = b_frag_gf32(Wq, 128, ks * 32, c0, lane, 128);
        Bk[ks] = b_frag_gf32(Wk, 128, ks * 32, c0, lane, 128);
        Bv[ks] = b_frag_gf32(Wv, 128, ks * 32, c0, lane, 128);
    }
    float bbq = bq[c0 + col], bbk = bk[c0 + col], bbv = bv[c0 + col];

#pragma unroll
    for (int st = 0; st < 4; ++st) {
        int rt = blockIdx.x * 4 + st;
        if (rt > NT - 1) rt = NT - 1;
        size_t base = (size_t)rt * 2048;
        __syncthreads();
        // async global -> LDS stage of the 16x128 f32 tile (8 KB, 16 B/lane x2)
        {
            int off = tid * 16;
            async_copy16((char*)xf + off,        (const char*)(x + base) + off);
            async_copy16((char*)xf + off + 4096, (const char*)(x + base) + off + 4096);
            async_wait0();
        }
        __syncthreads();
        ln_tile(xf, xbf, g1, be1, red, musig, tid);

        v8f aq = {}, ak = {}, av = {};
#pragma unroll
        for (int ks = 0; ks < 4; ++ks) {
            v16bf af = a_frag_lds(xbf, 128, ks * 32, lane);
            aq = wmma_bf16(af, Bq[ks], aq);
            ak = wmma_bf16(af, Bk[ks], ak);
            av = wmma_bf16(af, Bv[ks], av);
        }
#pragma unroll
        for (int r = 0; r < 8; ++r) {
            int row = r + half * 8;
            size_t idx = ((size_t)rt * 16 + row) * HH + c0 + col;
            qo[idx] = f2bf(aq[r] + bbq);
            ko[idx] = f2bf(ak[r] + bbk);
            vo[idx] = f2bf(av[r] + bbv);
        }
    }
}

// ============ K2: curv bias (WMMA) + q·k gather + segment max ============
__global__ __launch_bounds__(256) void k2_scores(
    const float* __restrict__ curv, const int* __restrict__ ei,
    const float* __restrict__ Wc, const float* __restrict__ bc,
    const unsigned short* __restrict__ q, const unsigned short* __restrict__ k,
    float* __restrict__ scores, float* smax) {
    __shared__ float bias[256][NHH];
    int tid = threadIdx.x, wave = tid >> 5, lane = tid & 31;
    size_t e0 = (size_t)blockIdx.x * 256;

    v16bf Bc[4];
#pragma unroll
    for (int ks = 0; ks < 4; ++ks) Bc[ks] = b_frag_gf32(Wc, NHH, ks * 32, 0, lane, NHH);

#pragma unroll
    for (int t = 0; t < 2; ++t) {
        size_t ebase = e0 + wave * 32 + t * 16;
        v8f acc = {};
#pragma unroll
        for (int ks = 0; ks < 4; ++ks)
            acc = wmma_bf16(a_frag_gf32(curv + ebase * HH, HH, ks * 32, lane), Bc[ks], acc);
        int col = lane & 15, half = lane >> 4;
        if (col < NHH) {
            float bb = bc[col];
#pragma unroll
            for (int r = 0; r < 8; ++r)
                bias[wave * 32 + t * 16 + r + half * 8][col] = acc[r] + bb;
        }
    }
    __syncthreads();

    size_t e = e0 + tid;
    int src = ei[e];
    int tgt = ei[(size_t)EE + e];
    const unsigned int* qp = (const unsigned int*)(q + (size_t)src * HH);
    const unsigned int* kp = (const unsigned int*)(k + (size_t)tgt * HH);
#pragma unroll
    for (int h = 0; h < NHH; ++h) {
        float acc = 0.f;
#pragma unroll
        for (int i = 0; i < 8; ++i) {
            unsigned int qa = qp[h * 8 + i], ka = kp[h * 8 + i];
            acc += bf2f((unsigned short)(qa & 0xffffu)) * bf2f((unsigned short)(ka & 0xffffu));
            acc += bf2f((unsigned short)(qa >> 16))     * bf2f((unsigned short)(ka >> 16));
        }
        float sc = acc * 0.25f + bias[tid][h];   // 1/sqrt(16)
        scores[e * NHH + h] = sc;
        atomicMaxF(&smax[(size_t)tgt * NHH + h], sc);
    }
}

// ============ K3: exp + segment sum ============
__global__ __launch_bounds__(256) void k3_exp(const int* __restrict__ ei, float* scores,
                                              const float* __restrict__ smax, float* ssum) {
    size_t e = (size_t)blockIdx.x * 256 + threadIdx.x;
    int tgt = ei[(size_t)EE + e];
#pragma unroll
    for (int h = 0; h < NHH; ++h) {
        float p = __expf(scores[e * NHH + h] - smax[(size_t)tgt * NHH + h]);
        scores[e * NHH + h] = p;
        atomicAdd(&ssum[(size_t)tgt * NHH + h], p);
    }
}

// ============ K4: weighted message scatter to src ============
__global__ __launch_bounds__(256) void k4_msg(const int* __restrict__ ei,
                                              const float* __restrict__ scores,
                                              const float* __restrict__ ssum,
                                              const unsigned short* __restrict__ v,
                                              float* outacc) {
    size_t t = (size_t)blockIdx.x * 256 + threadIdx.x;
    size_t e = t >> 3;
    int h = (int)(t & 7);
    int src = ei[e];
    int tgt = ei[(size_t)EE + e];
    float prob = scores[e * NHH + h] / ssum[(size_t)tgt * NHH + h];
    const unsigned short* vp = v + (size_t)tgt * HH + h * 16;
    float* op = outacc + (size_t)src * HH + h * 16;
#pragma unroll
    for (int d = 0; d < 16; ++d) atomicAdd(&op[d], prob * bf2f(vp[d]));
}

// ============ K5: out-proj + residual + LN2 + FFN, 4 row-tiles/block ============
__global__ __launch_bounds__(256) void k5_ffn(
    const float* __restrict__ x, const float* __restrict__ outacc,
    const float* __restrict__ Wo, const float* __restrict__ bo,
    const float* __restrict__ W1, const float* __restrict__ b1,
    const float* __restrict__ W2, const float* __restrict__ b2,
    const float* __restrict__ g2, const float* __restrict__ be2,
    float* __restrict__ y) {
    __shared__ __align__(16) unsigned short abf[2048];      // reused per subtile
    __shared__ float x2[4 * 2048];                          // 32 KB, persists
    __shared__ __align__(16) unsigned short xnbf[4 * 2048]; // 16 KB, persists
    __shared__ __align__(16) unsigned short h1[4 * 8192];   // 64 KB, persists
    __shared__ float red[512];
    __shared__ float musig[32];
    int tid = threadIdx.x, wave = tid >> 5, lane = tid & 31;
    int col = lane & 15, half = lane >> 4, c0 = wave * 16;

    v16bf BW[16];

    // ---- phase 1: Wo projection + residual + LN2, per subtile ----
#pragma unroll
    for (int ks = 0; ks < 4; ++ks) BW[ks] = b_frag_gf32(Wo, 128, ks * 32, c0, lane, 128);
    {
        float bb = bo[c0 + col];
#pragma unroll
        for (int st = 0; st < 4; ++st) {
            int rt = blockIdx.x * 4 + st;
            if (rt > NT - 1) rt = NT - 1;
            size_t base = (size_t)rt * 2048;
            __syncthreads();
#pragma unroll
            for (int j = 0; j < 8; ++j) abf[tid + 256 * j] = f2bf(outacc[base + tid + 256 * j]);
            __syncthreads();
            v8f acc = {};
#pragma unroll
            for (int ks = 0; ks < 4; ++ks)
                acc = wmma_bf16(a_frag_lds(abf, 128, ks * 32, lane), BW[ks], acc);
#pragma unroll
            for (int r = 0; r < 8; ++r) {
                int row = r + half * 8;
                x2[st * 2048 + row * 128 + c0 + col] = x[base + row * 128 + c0 + col] + acc[r] + bb;
            }
            __syncthreads();
            ln_tile(x2 + st * 2048, xnbf + st * 2048, g2, be2, red, musig, tid);
        }
    }

    // ---- phase 2: FFN1 (128 -> 512, relu) into LDS ----
#pragma unroll
    for (int tix = 0; tix < 4; ++tix) {
        int cc = (wave * 4 + tix) * 16;
#pragma unroll
        for (int ks = 0; ks < 4; ++ks) BW[tix * 4 + ks] = b_frag_gf32(W1, 512, ks * 32, cc, lane, 512);
    }
#pragma unroll
    for (int st = 0; st < 4; ++st) {
#pragma unroll
        for (int tix = 0; tix < 4; ++tix) {
            int cc = (wave * 4 + tix) * 16;
            v8f acc = {};
#pragma unroll
            for (int ks = 0; ks < 4; ++ks)
                acc = wmma_bf16(a_frag_lds(xnbf + st * 2048, 128, ks * 32, lane), BW[tix * 4 + ks], acc);
            float bb = b1[cc + col];
#pragma unroll
            for (int r = 0; r < 8; ++r) {
                int row = r + half * 8;
                h1[st * 8192 + row * 512 + cc + col] = f2bf(fmaxf(acc[r] + bb, 0.f));
            }
        }
    }
    __syncthreads();

    // ---- phase 3: FFN2 (512 -> 128) + residual -> output ----
#pragma unroll
    for (int ks = 0; ks < 16; ++ks) BW[ks] = b_frag_gf32(W2, 128, ks * 32, c0, lane, 128);
    {
        float bb = b2[c0 + col];
#pragma unroll
        for (int st = 0; st < 4; ++st) {
            int rt = blockIdx.x * 4 + st;
            if (rt > NT - 1) rt = NT - 1;
            size_t base = (size_t)rt * 2048;
            v8f acc = {};
#pragma unroll
            for (int ks = 0; ks < 16; ++ks)
                acc = wmma_bf16(a_frag_lds(h1 + st * 8192, 512, ks * 32, lane), BW[ks], acc);
#pragma unroll
            for (int r = 0; r < 8; ++r) {
                int row = r + half * 8;
                y[base + row * 128 + c0 + col] = x2[st * 2048 + row * 128 + c0 + col] + acc[r] + bb;
            }
        }
    }
}

extern "C" void kernel_launch(void* const* d_in, const int* in_sizes, int n_in,
                              void* d_out, int out_size, void* d_ws, size_t ws_size,
                              hipStream_t stream) {
    const float* x    = (const float*)d_in[0];
    const int*   ei   = (const int*)d_in[1];
    const float* curv = (const float*)d_in[2];
    const float* Wq = (const float*)d_in[3];  const float* bq = (const float*)d_in[4];
    const float* Wk = (const float*)d_in[5];  const float* bk = (const float*)d_in[6];
    const float* Wv = (const float*)d_in[7];  const float* bv = (const float*)d_in[8];
    const float* Wo = (const float*)d_in[9];  const float* bo = (const float*)d_in[10];
    const float* Wc = (const float*)d_in[11]; const float* bc = (const float*)d_in[12];
    const float* W1 = (const float*)d_in[13]; const float* b1 = (const float*)d_in[14];
    const float* W2 = (const float*)d_in[15]; const float* b2 = (const float*)d_in[16];
    const float* g1 = (const float*)d_in[17]; const float* be1 = (const float*)d_in[18];
    const float* g2 = (const float*)d_in[19]; const float* be2 = (const float*)d_in[20];
    float* y = (float*)d_out;

    // workspace layout (bytes)
    char* ws = (char*)d_ws;
    unsigned short* qw = (unsigned short*)(ws);                       // N*128 bf16
    unsigned short* kw = (unsigned short*)(ws + 12800000);            // N*128 bf16
    unsigned short* vw = (unsigned short*)(ws + 25600000);            // N*128 bf16
    float* scores      = (float*)(ws + 38400000);                     // E*8 f32
    float* smax        = (float*)(ws + 64000000);                     // N*8 f32
    float* ssum        = (float*)(ws + 65600000);                     // N*8 f32
    float* outacc      = (float*)(ws + 67200000);                     // N*128 f32

    dim3 blk(256);
    k0_init<<<dim3((NN * HH) / 256), blk, 0, stream>>>(smax, ssum, outacc);
    k1_ln_qkv<<<dim3(NB), blk, 0, stream>>>(x, Wq, bq, Wk, bk, Wv, bv, g1, be1, qw, kw, vw);
    k2_scores<<<dim3(EE / 256), blk, 0, stream>>>(curv, ei, Wc, bc, qw, kw, scores, smax);
    k3_exp<<<dim3(EE / 256), blk, 0, stream>>>(ei, scores, smax, ssum);
    k4_msg<<<dim3((EE * 8) / 256), blk, 0, stream>>>(ei, scores, ssum, vw, outacc);
    k5_ffn<<<dim3(NB), blk, 0, stream>>>(x, outacc, Wo, bo, W1, b1, W2, b2, g2, be2, y);
}
